// Block_24799141167816
// MI455X (gfx1250) — matrix operand order repaired and checked by reading, not compiled, not run on hardware
//
#include <hip/hip_runtime.h>
#include <hip/hip_bf16.h>

// ---------------------------------------------------------------------------
// Problem constants (from the reference)
// ---------------------------------------------------------------------------
#define BB    8
#define HH    56
#define WW2   56
#define CC    256
#define NHH   8
#define HDD   32
#define NN    3136            // HH*WW2
#define MM    (BB*NN)         // 25088 tokens (multiple of 128)
#define LL    9               // 3x3 window
#define PLL   49              // 7x7 pool
#define HID_  682
#define FC1N  1364            // 2*HID
#define FC1NP 1408            // padded to 128
#define FC2KP 704             // HID padded to 32
#define MPOOL 392             // B*49
#define MPOOLP 512            // padded to 128
#define TBLN  4096
#define NEGF  (-1e30f)

typedef unsigned short u16;
typedef __attribute__((ext_vector_type(16))) __bf16 v16bf;
typedef __attribute__((ext_vector_type(8)))  float  v8f;

// ---------------------------------------------------------------------------
// Helpers
// ---------------------------------------------------------------------------
__device__ __forceinline__ u16 f2bf(float f) {
    union { float f; unsigned u; } v; v.f = f;
    unsigned u = v.u;
    unsigned r = u + 0x7fffu + ((u >> 16) & 1u);   // round-to-nearest-even
    return (u16)(r >> 16);
}

__device__ __forceinline__ float gelu_f(float x) {
    return 0.5f * x * (1.0f + erff(x * 0.70710678118654752f));
}

// full-wave (32-lane) sum; every lane gets the total
__device__ __forceinline__ float wredsum(float v) {
#pragma unroll
    for (int o = 16; o > 0; o >>= 1) v += __shfl_xor(v, o, 32);
    return v;
}

// ---------------------------------------------------------------------------
// Weight convert + transpose: W[K,N] f32 -> Wt[Np,Kp] bf16 (zero padded)
// ---------------------------------------------------------------------------
__global__ void k_convT(const float* __restrict__ W, u16* __restrict__ Wt,
                        int K, int N, int Kp, int Np) {
    int idx = blockIdx.x * 256 + threadIdx.x;
    if (idx >= Np * Kp) return;
    int n = idx / Kp, k = idx % Kp;
    float v = (n < N && k < K) ? W[(size_t)k * N + n] : 0.0f;
    Wt[idx] = f2bf(v);
}

// ---------------------------------------------------------------------------
// Generic bf16 WMMA GEMM:  C[M,Np] = act( A[M,K](bf16) * Bt[Np,K]^T + bias )
// 128x128 tile per 256-thread block (8 waves, wave32).
// Wave (wm,wn) owns a 32x64 sub-tile: 2 A-frags x 4 B-frags = 8 WMMAs / K-step.
// act: 0 = none, 1 = exact GELU
// ---------------------------------------------------------------------------
__global__ __launch_bounds__(256) void k_gemm(
    const u16* __restrict__ A, const u16* __restrict__ Bt,
    const float* __restrict__ bias, int biasN,
    float* __restrict__ C, int Np, int K, int act)
{
    __shared__ __align__(16) u16 As[128 * 40];
    __shared__ __align__(16) u16 Bs[128 * 40];

    const int tid  = threadIdx.x;
    const int wv   = tid >> 5;          // 0..7  (wave32)
    const int lane = tid & 31;
    const int wm   = wv >> 1;           // 0..3  : 32-row group
    const int wn   = wv & 1;            // 0..1  : 64-col group
    const int m0   = blockIdx.y * 128;
    const int n0   = blockIdx.x * 128;
    const int lrow  = tid >> 1;         // 0..127
    const int lhalf = (tid & 1) * 16;   // 0 or 16 (elements)

    v8f acc[2][4];
#pragma unroll
    for (int i = 0; i < 2; ++i)
#pragma unroll
        for (int t = 0; t < 4; ++t)
#pragma unroll
            for (int e = 0; e < 8; ++e) acc[i][t][e] = 0.0f;

    const int mr  = lane & 15;
    const int akb = (lane >> 4) * 8;    // A frag K-sub-block (0/8)
    const int bkb = (lane >> 4) * 8;    // B frag K-sub-block (0/8)

    for (int kk = 0; kk < K; kk += 32) {
        const u16* ga = A  + (size_t)(m0 + lrow) * K + kk + lhalf;
        const u16* gb = Bt + (size_t)(n0 + lrow) * K + kk + lhalf;
        int4 a0 = ((const int4*)ga)[0], a1 = ((const int4*)ga)[1];
        int4 b0 = ((const int4*)gb)[0], b1 = ((const int4*)gb)[1];
        if (kk + 32 < K) {                    // global_prefetch_b8 of next slice
            __builtin_prefetch(ga + 32, 0, 0);
            __builtin_prefetch(gb + 32, 0, 0);
        }
        *(int4*)&As[lrow * 40 + lhalf]     = a0;
        *(int4*)&As[lrow * 40 + lhalf + 8] = a1;
        *(int4*)&Bs[lrow * 40 + lhalf]     = b0;
        *(int4*)&Bs[lrow * 40 + lhalf + 8] = b1;
        __syncthreads();

        union UV { int4 q[2]; v16bf v; } ua[2], ub;
        // A fragment: lanes 0-15 hold K {0..7, 16..23}; lanes 16-31 {8..15, 24..31}
#pragma unroll
        for (int i = 0; i < 2; ++i) {
            ua[i].q[0] = *(const int4*)&As[(wm * 32 + i * 16 + mr) * 40 + akb];
            ua[i].q[1] = *(const int4*)&As[(wm * 32 + i * 16 + mr) * 40 + akb + 16];
        }
#pragma unroll
        for (int t = 0; t < 4; ++t) {
            // B fragment: lanes 0-15 hold K {0..7, 16..23}; lanes 16-31 {8..15, 24..31}
            ub.q[0] = *(const int4*)&Bs[(wn * 64 + t * 16 + mr) * 40 + bkb];
            ub.q[1] = *(const int4*)&Bs[(wn * 64 + t * 16 + mr) * 40 + bkb + 16];
#pragma unroll
            for (int i = 0; i < 2; ++i)
                acc[i][t] = __builtin_amdgcn_wmma_f32_16x16x32_bf16(
                    false, ua[i].v, false, ub.v, (short)0, acc[i][t], false, false);
        }
        __syncthreads();
    }

    // C/D layout: lanes 0-15 -> N=lane, M=e ; lanes 16-31 -> N=lane-16, M=8+e
    const int nloc  = lane & 15;
    const int rbase = (lane >> 4) * 8;
#pragma unroll
    for (int i = 0; i < 2; ++i)
#pragma unroll
        for (int t = 0; t < 4; ++t) {
            int col = n0 + wn * 64 + t * 16 + nloc;
            float bv = (bias != nullptr && col < biasN) ? bias[col] : 0.0f;
#pragma unroll
            for (int e = 0; e < 8; ++e) {
                int row = m0 + wm * 32 + i * 16 + rbase + e;
                float c = acc[i][t][e] + bv;
                if (act == 1) c = gelu_f(c);
                C[(size_t)row * Np + col] = c;
            }
        }
}

// ---------------------------------------------------------------------------
// LayerNorm over C=256, one token per 256-thread block, bf16 output
// ---------------------------------------------------------------------------
__global__ __launch_bounds__(256) void k_ln(
    const float* __restrict__ X, const float* __restrict__ g,
    const float* __restrict__ b, u16* __restrict__ out_bf)
{
    __shared__ float sm[8];
    int row = blockIdx.x, c = threadIdx.x;
    int wv = c >> 5, ln = c & 31;
    size_t base = (size_t)row * CC;
    float x = X[base + c];
    float s = wredsum(x);
    if (ln == 0) sm[wv] = s;
    __syncthreads();
    float mean = (sm[0]+sm[1]+sm[2]+sm[3]+sm[4]+sm[5]+sm[6]+sm[7]) * (1.0f/256.0f);
    __syncthreads();
    float dx = x - mean;
    float s2 = wredsum(dx * dx);
    if (ln == 0) sm[wv] = s2;
    __syncthreads();
    float var = (sm[0]+sm[1]+sm[2]+sm[3]+sm[4]+sm[5]+sm[6]+sm[7]) * (1.0f/256.0f);
    out_bf[base + c] = f2bf(dx * rsqrtf(var + 1e-5f) * g[c] + b[c]);
}

// residual add + LayerNorm: Y = X + P (f32 out), YN = LN(Y) (bf16 out)
__global__ __launch_bounds__(256) void k_add_ln(
    const float* __restrict__ X, const float* __restrict__ P,
    const float* __restrict__ g, const float* __restrict__ b,
    float* __restrict__ Y, u16* __restrict__ YN)
{
    __shared__ float sm[8];
    int row = blockIdx.x, c = threadIdx.x;
    int wv = c >> 5, ln = c & 31;
    size_t base = (size_t)row * CC;
    float y = X[base + c] + P[base + c];
    Y[base + c] = y;
    float s = wredsum(y);
    if (ln == 0) sm[wv] = s;
    __syncthreads();
    float mean = (sm[0]+sm[1]+sm[2]+sm[3]+sm[4]+sm[5]+sm[6]+sm[7]) * (1.0f/256.0f);
    __syncthreads();
    float dx = y - mean;
    float s2 = wredsum(dx * dx);
    if (ln == 0) sm[wv] = s2;
    __syncthreads();
    float var = (sm[0]+sm[1]+sm[2]+sm[3]+sm[4]+sm[5]+sm[6]+sm[7]) * (1.0f/256.0f);
    YN[base + c] = f2bf(dx * rsqrtf(var + 1e-5f) * g[c] + b[c]);
}

// ---------------------------------------------------------------------------
// q/k post-processing: l2-norm, query embedding, softplus(temperature)*sls
// one wave per (token, head); lane = head-dim
// ---------------------------------------------------------------------------
__global__ __launch_bounds__(256) void k_qkv_post(
    const float* __restrict__ Q, const float* __restrict__ KV,
    const float* __restrict__ TEMP, const float* __restrict__ QE,
    const float* __restrict__ SLS,
    float* __restrict__ QN, float* __restrict__ QS, float* __restrict__ KN)
{
    int row = blockIdx.x;
    int h = threadIdx.x >> 5, d = threadIdx.x & 31;
    size_t qi = (size_t)row * CC + h * HDD + d;
    float qv = Q[qi];
    float qn = qv / fmaxf(sqrtf(wredsum(qv * qv)), 1e-12f);
    float sp = log1pf(expf(TEMP[h]));
    QN[qi] = qn;
    QS[qi] = (qn + QE[h * HDD + d]) * sp * SLS[0];
    float kv = KV[(size_t)row * (2 * CC) + h * HDD + d];
    KN[qi] = kv / fmaxf(sqrtf(wredsum(kv * kv)), 1e-12f);
}

// ---------------------------------------------------------------------------
// 8x8 average pool of xs + LayerNorm -> bf16 (padded to MPOOLP rows)
// ---------------------------------------------------------------------------
__global__ __launch_bounds__(256) void k_pool(
    const float* __restrict__ XS, const float* __restrict__ g,
    const float* __restrict__ b, u16* __restrict__ OUT)
{
    __shared__ float sm[8];
    int row = blockIdx.x, c = threadIdx.x;
    if (row >= MPOOL) { OUT[(size_t)row * CC + c] = 0; return; }
    int wv = c >> 5, ln = c & 31;
    int bi = row / PLL, cell = row % PLL;
    int pi = cell / 7, pj = cell % 7;
    float acc = 0.0f;
#pragma unroll 1
    for (int i = 0; i < 8; ++i)
        for (int j = 0; j < 8; ++j)
            acc += XS[((size_t)bi * NN + (pi * 8 + i) * WW2 + (pj * 8 + j)) * CC + c];
    float x = acc * (1.0f / 64.0f);
    float s = wredsum(x);
    if (ln == 0) sm[wv] = s;
    __syncthreads();
    float mean = (sm[0]+sm[1]+sm[2]+sm[3]+sm[4]+sm[5]+sm[6]+sm[7]) * (1.0f/256.0f);
    __syncthreads();
    float dx = x - mean;
    float s2 = wredsum(dx * dx);
    if (ln == 0) sm[wv] = s2;
    __syncthreads();
    float var = (sm[0]+sm[1]+sm[2]+sm[3]+sm[4]+sm[5]+sm[6]+sm[7]) * (1.0f/256.0f);
    OUT[(size_t)row * CC + c] = f2bf(dx * rsqrtf(var + 1e-5f) * g[c] + b[c]);
}

// l2-normalize k_pool -> KPN[(b*NH+h)*49 + m][32]
__global__ __launch_bounds__(256) void k_poolpost(
    const float* __restrict__ KVP, float* __restrict__ KPN)
{
    int row = blockIdx.x;                 // b*49 + m
    int h = threadIdx.x >> 5, d = threadIdx.x & 31;
    int bi = row / PLL, m = row % PLL;
    float v = KVP[(size_t)row * (2 * CC) + h * HDD + d];
    float nn = fmaxf(sqrtf(wredsum(v * v)), 1e-12f);
    KPN[(((size_t)bi * NHH + h) * PLL + m) * HDD + d] = v / nn;
}

// ---------------------------------------------------------------------------
// CPB MLP:  cpb[r][h] = relu(table[r]@W1 + b1) @ W2 + b2   (4096 x 8)
// ---------------------------------------------------------------------------
__global__ __launch_bounds__(512) void k_cpb(
    const float* __restrict__ TBL2, const float* __restrict__ W1,
    const float* __restrict__ B1, const float* __restrict__ W2,
    const float* __restrict__ B2, float* __restrict__ CPB)
{
    __shared__ float accs[8];
    int r = blockIdx.x, j = threadIdx.x;
    if (j < 8) accs[j] = B2[j];
    __syncthreads();
    float t0 = TBL2[r * 2], t1 = TBL2[r * 2 + 1];
    float hj = fmaxf(t0 * W1[j] + t1 * W1[512 + j] + B1[j], 0.0f);
#pragma unroll
    for (int h = 0; h < 8; ++h) atomicAdd(&accs[h], hj * W2[j * 8 + h]);
    __syncthreads();
    if (j < 8) CPB[(size_t)r * 8 + j] = accs[j];
}

// ---------------------------------------------------------------------------
// Fused attention core: local 3x3 scores + pooled 49 scores + softmax(58)
// + learnable-token correction + weighted sums -> bf16 out for proj GEMM.
// Block = 8 waves = 8 tokens of one (b,h); lane = head-dim.
// ---------------------------------------------------------------------------
__global__ __launch_bounds__(256) void k_attn(
    const float* __restrict__ QS, const float* __restrict__ QN,
    const float* __restrict__ KN, const float* __restrict__ KV,
    const float* __restrict__ KPN, const float* __restrict__ KVP,
    const float* __restrict__ CPB, const int* __restrict__ RPI,
    const float* __restrict__ RPB, const float* __restrict__ LT,
    const float* __restrict__ LB, u16* __restrict__ OUT)
{
    __shared__ float kpl[PLL * HDD];
    __shared__ float vpl[PLL * HDD];
    __shared__ float ltl[HDD * LL];
    __shared__ float rpbl[LL], lbl[LL];

    int bid = blockIdx.x;
    int grp = bid % (NN / 8);
    int bh  = bid / (NN / 8);
    int h = bh % NHH, bi = bh / NHH;
    int tid = threadIdx.x;

    for (int i = tid; i < PLL * HDD; i += 256) {
        kpl[i] = KPN[(((size_t)bi * NHH + h) * PLL) * HDD + i];
        int m = i >> 5, d2 = i & 31;
        vpl[i] = KVP[((size_t)bi * PLL + m) * (2 * CC) + CC + h * HDD + d2];
    }
    for (int i = tid; i < HDD * LL; i += 256) ltl[i] = LT[h * HDD * LL + i];
    if (tid < LL) { rpbl[tid] = RPB[h * LL + tid]; lbl[tid] = LB[h * LL + tid]; }
    __syncthreads();

    int wv = tid >> 5, d = tid & 31;
    int tok = grp * 8 + wv;                     // token within image
    int ii0 = tok / WW2, jj0 = tok % WW2;
    size_t row = (size_t)bi * NN + tok;
    float qs = QS[row * CC + h * HDD + d];
    float qn = QN[row * CC + h * HDD + d];

    float aloc[LL];
#pragma unroll
    for (int l = 0; l < LL; ++l) {
        int di = l / 3 - 1, dj = l % 3 - 1;
        int ii = ii0 + di, jj = jj0 + dj;
        if (ii >= 0 && ii < HH && jj >= 0 && jj < WW2) {
            size_t nb = (size_t)bi * NN + ii * WW2 + jj;
            aloc[l] = wredsum(qs * KN[nb * CC + h * HDD + d]) + rpbl[l];
        } else aloc[l] = NEGF;
    }

    float apool[PLL];
#pragma unroll
    for (int m = 0; m < PLL; ++m) {
        float s = wredsum(qs * kpl[m * HDD + d]);
        apool[m] = s + CPB[(size_t)RPI[tok * PLL + m] * 8 + h];
    }

    // softmax over 58 logits
    float mx = NEGF;
#pragma unroll
    for (int l = 0; l < LL; ++l) mx = fmaxf(mx, aloc[l]);
#pragma unroll
    for (int m = 0; m < PLL; ++m) mx = fmaxf(mx, apool[m]);
    float sum = 0.0f;
#pragma unroll
    for (int l = 0; l < LL; ++l) { aloc[l] = expf(aloc[l] - mx); sum += aloc[l]; }
#pragma unroll
    for (int m = 0; m < PLL; ++m) { apool[m] = expf(apool[m] - mx); sum += apool[m]; }
    float inv = 1.0f / sum;

    float xl = 0.0f, xp = 0.0f;
#pragma unroll
    for (int l = 0; l < LL; ++l) {
        float ex = wredsum(qn * ltl[d * LL + l]) + lbl[l];
        float al = aloc[l] * inv + ex;
        int di = l / 3 - 1, dj = l % 3 - 1;
        int ii = ii0 + di, jj = jj0 + dj;
        if (ii >= 0 && ii < HH && jj >= 0 && jj < WW2) {
            size_t nb = (size_t)bi * NN + ii * WW2 + jj;
            xl += al * KV[nb * (2 * CC) + CC + h * HDD + d];
        }
    }
#pragma unroll
    for (int m = 0; m < PLL; ++m) xp += (apool[m] * inv) * vpl[m * HDD + d];

    OUT[row * CC + h * HDD + d] = f2bf(xl + xp);
}

// ---------------------------------------------------------------------------
// GLU: depthwise 3x3 on 'a' half of fc1 output, exact-GELU gate -> bf16
// ---------------------------------------------------------------------------
__global__ __launch_bounds__(256) void k_glu(
    const float* __restrict__ U, const float* __restrict__ DWW,
    const float* __restrict__ DWB, u16* __restrict__ GLU)
{
    int row = blockIdx.x;
    int bi = row / NN, tok = row % NN;
    int ii0 = tok / WW2, jj0 = tok % WW2;
    for (int c = threadIdx.x; c < FC2KP; c += 256) {
        if (c < HID_) {
            float acc = DWB[c];
#pragma unroll
            for (int ki = 0; ki < 3; ++ki)
#pragma unroll
                for (int kj = 0; kj < 3; ++kj) {
                    int ii = ii0 + ki - 1, jj = jj0 + kj - 1;
                    if (ii >= 0 && ii < HH && jj >= 0 && jj < WW2)
                        acc += U[((size_t)bi * NN + ii * WW2 + jj) * FC1NP + c]
                             * DWW[c * 9 + ki * 3 + kj];
                }
            float gate = U[(size_t)row * FC1NP + HID_ + c];
            GLU[(size_t)row * FC2KP + c] = f2bf(gelu_f(acc) * gate);
        } else {
            GLU[(size_t)row * FC2KP + c] = 0;
        }
    }
}

// final residual
__global__ __launch_bounds__(256) void k_final(
    const float* __restrict__ Y, const float* __restrict__ F2,
    float* __restrict__ OUT)
{
    size_t idx = (size_t)blockIdx.x * 256 + threadIdx.x;
    OUT[idx] = Y[idx] + F2[idx];
}

// ---------------------------------------------------------------------------
// Host launch
// ---------------------------------------------------------------------------
extern "C" void kernel_launch(void* const* d_in, const int* in_sizes, int n_in,
                              void* d_out, int out_size, void* d_ws, size_t ws_size,
                              hipStream_t stream) {
    // inputs: x, then the 29 params flattened in alphabetical pytree-key order,
    // then relative_pos_index, relative_coords_table, seq_length_scale, H, W
    const float* x      = (const float*)d_in[0];
    const float* cpb1_b = (const float*)d_in[1];
    const float* cpb1_w = (const float*)d_in[2];
    const float* cpb2_b = (const float*)d_in[3];
    const float* cpb2_w = (const float*)d_in[4];
    const float* dw_b   = (const float*)d_in[5];
    const float* dw_w   = (const float*)d_in[6];
    const float* fc1_b  = (const float*)d_in[7];
    const float* fc1_w  = (const float*)d_in[8];
    const float* fc2_b  = (const float*)d_in[9];
    const float* fc2_w  = (const float*)d_in[10];
    const float* kv_b   = (const float*)d_in[11];
    const float* kv_w   = (const float*)d_in[12];
    const float* lbias  = (const float*)d_in[13];   // learnable_bias
    const float* ltok   = (const float*)d_in[14];   // learnable_tokens
    const float* n1_b   = (const float*)d_in[15];
    const float* n1_g   = (const float*)d_in[16];
    const float* n2_b   = (const float*)d_in[17];
    const float* n2_g   = (const float*)d_in[18];
    const float* np_b   = (const float*)d_in[19];   // norm_b (pool LN)
    const float* np_g   = (const float*)d_in[20];   // norm_g
    const float* proj_b = (const float*)d_in[21];
    const float* proj_w = (const float*)d_in[22];
    const float* q_b    = (const float*)d_in[23];
    const float* q_w    = (const float*)d_in[24];
    const float* qemb   = (const float*)d_in[25];   // query_embedding
    const float* rpb    = (const float*)d_in[26];   // rpb_local
    const float* sr_b   = (const float*)d_in[27];
    const float* sr_w   = (const float*)d_in[28];
    const float* temp_p = (const float*)d_in[29];   // temperature
    const int*   rpi    = (const int*)d_in[30];     // relative_pos_index
    const float* table  = (const float*)d_in[31];   // relative_coords_table
    const float* sls    = (const float*)d_in[32];   // seq_length_scale
    (void)in_sizes; (void)n_in; (void)out_size; (void)ws_size;

    // ---- workspace layout (256B aligned) ----
    char* ws = (char*)d_ws;
    size_t off = 0;
    auto take = [&](size_t bytes) {
        size_t o = off; off = (off + bytes + 255) & ~(size_t)255; return o;
    };
    u16*   xn    = (u16*)  (ws + take((size_t)MM * CC * 2));        // LN1(x) bf16
    u16*   wqT   = (u16*)  (ws + take((size_t)CC * CC * 2));
    u16*   wkvT  = (u16*)  (ws + take((size_t)2 * CC * CC * 2));
    u16*   wsrT  = (u16*)  (ws + take((size_t)CC * CC * 2));
    u16*   wprT  = (u16*)  (ws + take((size_t)CC * CC * 2));
    u16*   wf1T  = (u16*)  (ws + take((size_t)FC1NP * CC * 2));
    u16*   wf2T  = (u16*)  (ws + take((size_t)CC * FC2KP * 2));
    float* qf    = (float*)(ws + take((size_t)MM * CC * 4));
    float* kvf   = (float*)(ws + take((size_t)MM * 2 * CC * 4));
    float* qnb   = (float*)(ws + take((size_t)MM * CC * 4));
    float* qsb   = (float*)(ws + take((size_t)MM * CC * 4));
    float* knb   = (float*)(ws + take((size_t)MM * CC * 4));
    float* xsb   = (float*)(ws + take((size_t)MM * CC * 4));
    u16*   xpl   = (u16*)  (ws + take((size_t)MPOOLP * CC * 2));
    float* kvp   = (float*)(ws + take((size_t)MPOOLP * 2 * CC * 4));
    float* kpn   = (float*)(ws + take((size_t)BB * NHH * PLL * HDD * 4));
    float* cpb   = (float*)(ws + take((size_t)TBLN * 8 * 4));
    u16*   aout  = (u16*)  (ws + take((size_t)MM * CC * 2));
    float* projo = (float*)(ws + take((size_t)MM * CC * 4));
    float* yb    = (float*)(ws + take((size_t)MM * CC * 4));
    u16*   ynb   = (u16*)  (ws + take((size_t)MM * CC * 2));
    float* ub    = (float*)(ws + take((size_t)MM * FC1NP * 4));
    u16*   glub  = (u16*)  (ws + take((size_t)MM * FC2KP * 2));
    float* f2o   = (float*)(ws + take((size_t)MM * CC * 4));

    auto gemm = [&](const u16* A, const u16* Bt, const float* bias, int biasN,
                    float* C, int M, int Np, int K, int act) {
        dim3 g((unsigned)(Np / 128), (unsigned)(M / 128));
        k_gemm<<<g, 256, 0, stream>>>(A, Bt, bias, biasN, C, Np, K, act);
    };
    auto convT = [&](const float* W, u16* Wt, int K, int N, int Kp, int Np) {
        int tot = Np * Kp;
        k_convT<<<(tot + 255) / 256, 256, 0, stream>>>(W, Wt, K, N, Kp, Np);
    };

    // 1) weights -> bf16 transposed (padded)
    convT(q_w,    wqT,  CC, CC,       CC,    CC);
    convT(kv_w,   wkvT, CC, 2 * CC,   CC,    2 * CC);
    convT(sr_w,   wsrT, CC, CC,       CC,    CC);
    convT(proj_w, wprT, CC, CC,       CC,    CC);
    convT(fc1_w,  wf1T, CC, FC1N,     CC,    FC1NP);
    convT(fc2_w,  wf2T, HID_, CC,     FC2KP, CC);

    // 2) xn = LN1(x)
    k_ln<<<MM, 256, 0, stream>>>(x, n1_g, n1_b, xn);

    // 3) big GEMMs off xn: q, kv, xs = gelu(xn@sr_w + sr_b)
    gemm(xn, wqT,  q_b,  CC,     qf,  MM, CC,     CC, 0);
    gemm(xn, wkvT, kv_b, 2 * CC, kvf, MM, 2 * CC, CC, 0);
    gemm(xn, wsrT, sr_b, CC,     xsb, MM, CC,     CC, 1);

    // 4) q/k normalization + scaling
    k_qkv_post<<<MM, 256, 0, stream>>>(qf, kvf, temp_p, qemb, sls, qnb, qsb, knb);

    // 5) pooled path: avg-pool + LN -> bf16, kv_pool GEMM, l2-norm k_pool
    k_pool<<<MPOOLP, 256, 0, stream>>>(xsb, np_g, np_b, xpl);
    gemm(xpl, wkvT, kv_b, 2 * CC, kvp, MPOOLP, 2 * CC, CC, 0);
    k_poolpost<<<MPOOL, 256, 0, stream>>>(kvp, kpn);

    // 6) CPB relative-position bias table
    k_cpb<<<TBLN, 512, 0, stream>>>(table, cpb1_w, cpb1_b, cpb2_w, cpb2_b, cpb);

    // 7) fused attention
    k_attn<<<BB * NHH * (NN / 8), 256, 0, stream>>>(
        qsb, qnb, knb, kvf, kpn, kvp, cpb, rpi, rpb, ltok, lbias, aout);

    // 8) proj + residual + LN2
    gemm(aout, wprT, proj_b, CC, projo, MM, CC, CC, 0);
    k_add_ln<<<MM, 256, 0, stream>>>(x, projo, n2_g, n2_b, yb, ynb);

    // 9) GLU MLP: fc1, depthwise+gate, fc2
    gemm(ynb, wf1T, fc1_b, FC1N, ub, MM, FC1NP, CC, 0);
    k_glu<<<MM, 256, 0, stream>>>(ub, dw_w, dw_b, glub);
    gemm(glub, wf2T, fc2_b, CC, f2o, MM, CC, FC2KP, 0);

    // 10) final residual -> d_out
    k_final<<<MM, 256, 0, stream>>>(yb, f2o, (float*)d_out);
}
